// ImprovedDIN_44014824849515
// MI455X (gfx1250) — compile-verified, matrix operand order checked by reading, not run on hardware
//
#include <hip/hip_runtime.h>
#include <hip/hip_bf16.h>
#include <stdint.h>

typedef __attribute__((ext_vector_type(16))) _Float16 v16h;
typedef __attribute__((ext_vector_type(8)))  float    v8f;

union Frag { v16h v; uint4 q[2]; };

#define B_SZ 8192
#define L_SZ 200
#define D_SZ 64

// ---------------- LDS layout for the attention kernel (dynamic) ----------------
#define OFF_KF32   0          // 200*64 f32                = 51200
#define OFF_KF16   51200      // 208*64 f16                = 26624
#define OFF_H1     77824      // 208*64 f16                = 26624
#define OFF_H2     104448     // 208*32 f16                = 13312
#define OFF_W1T    117760     // 64*64 f16 (W1k^T [n][k])  = 8192
#define OFF_W2T    125952     // 32*64 f16 (W2^T  [n][k])  = 4096
#define OFF_BIAS1  130048     // 64 f32 (qW1 + b1)
#define OFF_BIAS2  130304     // 32 f32
#define OFF_W3     130432     // 32 f32
#define OFF_QV     130560     // 64 f32
#define OFF_SCORES 130816     // 208 f32
#define OFF_IDX    131648     // 200 i32
#define OFF_RED    132448     // 256 f32
#define ATT_SMEM   133472

__global__ __launch_bounds__(256)
void din_attention_kernel(const int* __restrict__ ad, const int* __restrict__ brand,
                          const int* __restrict__ cate,
                          const int* __restrict__ hist_ads, const int* __restrict__ hist_brands,
                          const int* __restrict__ hist_cates,
                          const float* __restrict__ E_ad, const float* __restrict__ E_brand,
                          const float* __restrict__ E_cate,
                          const float* __restrict__ attW1, const float* __restrict__ attb1,
                          const float* __restrict__ attb2,
                          const float* __restrict__ attW3, const float* __restrict__ attb3,
                          const _Float16* __restrict__ aW1Tg,  // [64][64] f16, pre-transposed
                          const _Float16* __restrict__ aW2Tg,  // [32][64] f16, pre-transposed
                          float* __restrict__ hv)
{
    extern __shared__ __align__(16) char smem[];
    float*     kf32   = (float*)(smem + OFF_KF32);
    _Float16*  kf16   = (_Float16*)(smem + OFF_KF16);
    _Float16*  h1     = (_Float16*)(smem + OFF_H1);
    _Float16*  h2     = (_Float16*)(smem + OFF_H2);
    _Float16*  w1T    = (_Float16*)(smem + OFF_W1T);
    _Float16*  w2T    = (_Float16*)(smem + OFF_W2T);
    float*     bias1  = (float*)(smem + OFF_BIAS1);
    float*     bias2  = (float*)(smem + OFF_BIAS2);
    float*     w3     = (float*)(smem + OFF_W3);
    float*     qv     = (float*)(smem + OFF_QV);
    float*     scores = (float*)(smem + OFF_SCORES);
    int*       idxbuf = (int*)(smem + OFF_IDX);
    float*     red    = (float*)(smem + OFF_RED);

    const int t   = threadIdx.x;
    const int b   = blockIdx.x;
    const int att = blockIdx.y;

    const int* qidx; const int* hrow; const float* tab;
    if (att == 0)      { qidx = ad;    hrow = hist_ads;    tab = E_ad;    }
    else if (att == 1) { qidx = brand; hrow = hist_brands; tab = E_brand; }
    else               { qidx = cate;  hrow = hist_cates;  tab = E_cate;  }
    hrow += (size_t)b * L_SZ;

    // stage 0: history indices into LDS
    if (t < L_SZ) idxbuf[t] = hrow[t];
    __syncthreads();

    // stage 1: async gather of 200 rows (16 x 16B chunks per 256B row) straight
    // into LDS via the CDNA5 async global->LDS path (no VGPR round-trip).
    {
        unsigned kbase = (unsigned)(size_t)(void*)kf32;   // low 32 bits = LDS offset
        for (int chunk = t; chunk < L_SZ * 16; chunk += 256) {
            int row = chunk >> 4, sub = chunk & 15;
            uint64_t ga = (uint64_t)((const char*)(tab + (size_t)idxbuf[row] * D_SZ) + sub * 16);
            unsigned  la = kbase + (unsigned)(row * 256 + sub * 16);
            asm volatile("global_load_async_to_lds_b128 %0, %1, off"
                         :: "v"(la), "v"(ga) : "memory");
        }
    }

    // overlap while async copies fly: query embedding + f16 weight staging
    if (t < 64) qv[t] = tab[(size_t)qidx[b] * D_SZ + t];
    {
        const uint4* s1 = (const uint4*)aW1Tg;            // 8192 B = 512 x 16B
        uint4*       d1 = (uint4*)w1T;
        for (int e = t; e < 512; e += 256) d1[e] = s1[e];
        const uint4* s2 = (const uint4*)aW2Tg;            // 4096 B = 256 x 16B
        uint4*       d2 = (uint4*)w2T;
        if (t < 256) d2[t] = s2[t];
    }
    if (t < 32) { w3[t] = attW3[t]; bias2[t] = attb2[t]; }

    asm volatile("s_wait_asynccnt 0" ::: "memory");
    __syncthreads();

    // stage 2: f32 -> f16 for WMMA A-operands; zero-pad rows 200..207
    for (int e = t; e < L_SZ * D_SZ; e += 256) kf16[e] = (_Float16)kf32[e];
    for (int e = L_SZ * D_SZ + t; e < 208 * 64; e += 256) kf16[e] = (_Float16)0.f;
    // collapsed q-half of layer1: bias1[n] = b1[n] + sum_d q[d] * W1[d][n]
    if (t < 64) {
        float s = attb1[t];
        for (int d = 0; d < 64; ++d) s += qv[d] * attW1[d * 64 + t];
        bias1[t] = s;
    }
    __syncthreads();

    const int wv = t >> 5, lane = t & 31;
    const int mrow = lane & 15, kh = lane >> 4;

    // ---------------- GEMM1: H1[208,64] = relu(K[208,64] @ W1k[64,64] + bias1)
    {
        Frag bf[4][2];
        #pragma unroll
        for (int nt = 0; nt < 4; ++nt)
            #pragma unroll
            for (int ks = 0; ks < 2; ++ks) {
                const _Float16* bp = w1T + (nt * 16 + mrow) * 64 + ks * 32 + kh * 8;
                bf[nt][ks].q[0] = *(const uint4*)bp;
                bf[nt][ks].q[1] = *(const uint4*)(bp + 16);
            }
        for (int mt = wv; mt < 13; mt += 8) {
            Frag a0, a1;
            const _Float16* ap = kf16 + (mt * 16 + mrow) * 64 + kh * 8;
            a0.q[0] = *(const uint4*)ap;        a0.q[1] = *(const uint4*)(ap + 16);
            a1.q[0] = *(const uint4*)(ap + 32); a1.q[1] = *(const uint4*)(ap + 48);
            #pragma unroll
            for (int nt = 0; nt < 4; ++nt) {
                v8f c = {};
                c = __builtin_amdgcn_wmma_f32_16x16x32_f16(false, a0.v, false, bf[nt][0].v,
                                                           (short)0, c, false, false);
                c = __builtin_amdgcn_wmma_f32_16x16x32_f16(false, a1.v, false, bf[nt][1].v,
                                                           (short)0, c, false, false);
                int ncol = nt * 16 + mrow;
                float bs = bias1[ncol];
                #pragma unroll
                for (int r = 0; r < 8; ++r) {
                    float vv = c[r] + bs; vv = vv > 0.f ? vv : 0.f;
                    h1[(mt * 16 + r + 8 * kh) * 64 + ncol] = (_Float16)vv;
                }
            }
        }
    }
    __syncthreads();

    // ---------------- GEMM2: H2[208,32] = relu(H1 @ W2[64,32] + b2)
    {
        Frag bf[2][2];
        #pragma unroll
        for (int nt = 0; nt < 2; ++nt)
            #pragma unroll
            for (int ks = 0; ks < 2; ++ks) {
                const _Float16* bp = w2T + (nt * 16 + mrow) * 64 + ks * 32 + kh * 8;
                bf[nt][ks].q[0] = *(const uint4*)bp;
                bf[nt][ks].q[1] = *(const uint4*)(bp + 16);
            }
        for (int mt = wv; mt < 13; mt += 8) {
            Frag a0, a1;
            const _Float16* ap = h1 + (mt * 16 + mrow) * 64 + kh * 8;
            a0.q[0] = *(const uint4*)ap;        a0.q[1] = *(const uint4*)(ap + 16);
            a1.q[0] = *(const uint4*)(ap + 32); a1.q[1] = *(const uint4*)(ap + 48);
            #pragma unroll
            for (int nt = 0; nt < 2; ++nt) {
                v8f c = {};
                c = __builtin_amdgcn_wmma_f32_16x16x32_f16(false, a0.v, false, bf[nt][0].v,
                                                           (short)0, c, false, false);
                c = __builtin_amdgcn_wmma_f32_16x16x32_f16(false, a1.v, false, bf[nt][1].v,
                                                           (short)0, c, false, false);
                int ncol = nt * 16 + mrow;
                float bs = bias2[ncol];
                #pragma unroll
                for (int r = 0; r < 8; ++r) {
                    float vv = c[r] + bs; vv = vv > 0.f ? vv : 0.f;
                    h2[(mt * 16 + r + 8 * kh) * 32 + ncol] = (_Float16)vv;
                }
            }
        }
    }
    __syncthreads();

    // ---------------- scores + softmax over L=200
    if (t < L_SZ) {
        float s = attb3[0];
        const _Float16* hp = h2 + t * 32;
        #pragma unroll
        for (int j = 0; j < 32; ++j) s += (float)hp[j] * w3[j];
        scores[t] = s;
    }
    __syncthreads();
    red[t] = (t < L_SZ) ? scores[t] : -1e30f;
    __syncthreads();
    for (int s2 = 128; s2 > 0; s2 >>= 1) {
        if (t < s2) red[t] = fmaxf(red[t], red[t + s2]);
        __syncthreads();
    }
    float mx = red[0];
    __syncthreads();
    float ev = 0.f;
    if (t < L_SZ) { ev = __expf(scores[t] - mx); scores[t] = ev; }
    red[t] = ev;
    __syncthreads();
    for (int s2 = 128; s2 > 0; s2 >>= 1) {
        if (t < s2) red[t] += red[t + s2];
        __syncthreads();
    }
    float inv = 1.f / red[0];
    __syncthreads();
    if (t < L_SZ) scores[t] *= inv;
    __syncthreads();

    // ---------------- weighted sum (f32 accumulate from the f32 LDS copy)
    {
        int d = t & 63, g = t >> 6;                 // 4 groups of 50 rows
        float acc = 0.f;
        for (int l = g * 50; l < g * 50 + 50; ++l) acc += scores[l] * kf32[l * 64 + d];
        red[t] = acc;
        __syncthreads();
        if (t < 64) {
            float s = red[t] + red[t + 64] + red[t + 128] + red[t + 192];
            hv[((size_t)att * B_SZ + b) * 64 + t] = s;
        }
    }
}

// --------- prep: transpose + f16-convert all GEMM weights into workspace ------
#define PREP_N1 (128 * 512)                 // head W1
#define PREP_N2 (PREP_N1 + 64 * 128)        // head W2
#define PREP_N3 (PREP_N2 + 64 * 64)         // att W1 (k-half)
#define PREP_N4 (PREP_N3 + 32 * 64)         // att W2
__global__ __launch_bounds__(256)
void din_prep_kernel(const float* __restrict__ W1, const float* __restrict__ W2,
                     const float* __restrict__ aW1, const float* __restrict__ aW2,
                     _Float16* __restrict__ W1T, _Float16* __restrict__ W2T,
                     _Float16* __restrict__ aW1T, _Float16* __restrict__ aW2T)
{
    int i = blockIdx.x * 256 + threadIdx.x;
    if (i < PREP_N1) {
        int n = i >> 9, k = i & 511;
        W1T[n * 512 + k] = (_Float16)W1[k * 128 + n];        // mlp_W1 is [512][128]
    } else if (i < PREP_N2) {
        int j = i - PREP_N1, n = j >> 7, k = j & 127;
        W2T[n * 128 + k] = (_Float16)W2[k * 64 + n];         // mlp_W2 is [128][64]
    } else if (i < PREP_N3) {
        int j = i - PREP_N2, n = j >> 6, k = j & 63;
        aW1T[n * 64 + k] = (_Float16)aW1[(64 + k) * 64 + n]; // att_W1 is [128][64]
    } else if (i < PREP_N4) {
        int j = i - PREP_N3, n = j >> 6, k = j & 63;
        aW2T[n * 64 + k] = (_Float16)aW2[k * 32 + n];        // att_W2 is [64][32]
    }
}

// ----------------------- head MLP: 16 batch rows / block ----------------------
__global__ __launch_bounds__(256)
void din_head_kernel(const int* __restrict__ user, const int* __restrict__ ad,
                     const int* __restrict__ brand, const int* __restrict__ cate,
                     const int* __restrict__ camp,
                     const float* __restrict__ E_user, const float* __restrict__ E_ad,
                     const float* __restrict__ E_brand, const float* __restrict__ E_cate,
                     const float* __restrict__ E_camp,
                     const float* __restrict__ hv,
                     const _Float16* __restrict__ W1T, const float* __restrict__ b1g,
                     const _Float16* __restrict__ W2T, const float* __restrict__ b2g,
                     const float* __restrict__ W3g, const float* __restrict__ b3g,
                     float* __restrict__ out)
{
    __shared__ _Float16 xf[16 * 512];
    __shared__ _Float16 h1[16 * 128];
    __shared__ _Float16 h2[16 * 64];

    const int t  = threadIdx.x;
    const int b0 = blockIdx.x * 16;

    // build x = concat(u,a,b,c,p,hv_ads,hv_brands,hv_cates) as f16 [16][512]
    for (int e = t; e < 16 * 512; e += 256) {
        int row = e >> 9, col = e & 511, seg = col >> 6, d = col & 63;
        int bi = b0 + row;
        const float* src;
        switch (seg) {
            case 0:  src = E_user  + (size_t)user[bi]  * 64; break;
            case 1:  src = E_ad    + (size_t)ad[bi]    * 64; break;
            case 2:  src = E_brand + (size_t)brand[bi] * 64; break;
            case 3:  src = E_cate  + (size_t)cate[bi]  * 64; break;
            case 4:  src = E_camp  + (size_t)camp[bi]  * 64; break;
            default: src = hv + ((size_t)(seg - 5) * B_SZ + bi) * 64; break;
        }
        xf[e] = (_Float16)src[d];
    }
    __syncthreads();

    const int wv = t >> 5, lane = t & 31, mrow = lane & 15, kh = lane >> 4;

    // layer1: [16,512] @ [512,128]; one N-tile per wave, K = 16 steps of 32
    {
        const int nt = wv;
        v8f c = {};
        for (int ks = 0; ks < 16; ++ks) {
            Frag a, bf;
            const _Float16* ap = xf + mrow * 512 + ks * 32 + kh * 8;
            a.q[0] = *(const uint4*)ap;  a.q[1] = *(const uint4*)(ap + 16);
            const _Float16* bp = W1T + (size_t)(nt * 16 + mrow) * 512 + ks * 32 + kh * 8;
            bf.q[0] = *(const uint4*)bp; bf.q[1] = *(const uint4*)(bp + 16);
            c = __builtin_amdgcn_wmma_f32_16x16x32_f16(false, a.v, false, bf.v,
                                                       (short)0, c, false, false);
        }
        int ncol = nt * 16 + mrow;
        float bs = b1g[ncol];
        #pragma unroll
        for (int r = 0; r < 8; ++r) {
            float vv = c[r] + bs; vv = vv > 0.f ? vv : 0.f;
            h1[(r + 8 * kh) * 128 + ncol] = (_Float16)vv;
        }
    }
    __syncthreads();

    // layer2: [16,128] @ [128,64]; waves 0..3, K = 4 steps
    if (wv < 4) {
        const int nt = wv;
        v8f c = {};
        for (int ks = 0; ks < 4; ++ks) {
            Frag a, bf;
            const _Float16* ap = h1 + mrow * 128 + ks * 32 + kh * 8;
            a.q[0] = *(const uint4*)ap;  a.q[1] = *(const uint4*)(ap + 16);
            const _Float16* bp = W2T + (size_t)(nt * 16 + mrow) * 128 + ks * 32 + kh * 8;
            bf.q[0] = *(const uint4*)bp; bf.q[1] = *(const uint4*)(bp + 16);
            c = __builtin_amdgcn_wmma_f32_16x16x32_f16(false, a.v, false, bf.v,
                                                       (short)0, c, false, false);
        }
        int ncol = nt * 16 + mrow;
        float bs = b2g[ncol];
        #pragma unroll
        for (int r = 0; r < 8; ++r) {
            float vv = c[r] + bs; vv = vv > 0.f ? vv : 0.f;
            h2[(r + 8 * kh) * 64 + ncol] = (_Float16)vv;
        }
    }
    __syncthreads();

    // layer3: 64-dot per row
    if (t < 16) {
        float s = b3g[0];
        #pragma unroll
        for (int j = 0; j < 64; ++j) s += (float)h2[t * 64 + j] * W3g[j];
        out[b0 + t] = s;
    }
}

extern "C" void kernel_launch(void* const* d_in, const int* in_sizes, int n_in,
                              void* d_out, int out_size, void* d_ws, size_t ws_size,
                              hipStream_t stream) {
    const int* user  = (const int*)d_in[0];
    const int* ad    = (const int*)d_in[1];
    const int* brand = (const int*)d_in[2];
    const int* cate  = (const int*)d_in[3];
    const int* camp  = (const int*)d_in[4];
    const int* hist_ads    = (const int*)d_in[5];
    const int* hist_brands = (const int*)d_in[6];
    const int* hist_cates  = (const int*)d_in[7];
    const float* E_ad    = (const float*)d_in[8];
    const float* E_user  = (const float*)d_in[9];
    const float* E_brand = (const float*)d_in[10];
    const float* E_cate  = (const float*)d_in[11];
    const float* E_camp  = (const float*)d_in[12];
    const float* attW1 = (const float*)d_in[13]; const float* attb1 = (const float*)d_in[14];
    const float* attW2 = (const float*)d_in[15]; const float* attb2 = (const float*)d_in[16];
    const float* attW3 = (const float*)d_in[17]; const float* attb3 = (const float*)d_in[18];
    const float* mlpW1 = (const float*)d_in[19]; const float* mlpb1 = (const float*)d_in[20];
    const float* mlpW2 = (const float*)d_in[21]; const float* mlpb2 = (const float*)d_in[22];
    const float* mlpW3 = (const float*)d_in[23]; const float* mlpb3 = (const float*)d_in[24];

    char* ws = (char*)d_ws;
    _Float16* W1T  = (_Float16*)ws;                      // 128*512*2 = 131072 B
    _Float16* W2T  = (_Float16*)(ws + 131072);           // 64*128*2  = 16384 B
    _Float16* aW1T = (_Float16*)(ws + 147456);           // 64*64*2   = 8192 B
    _Float16* aW2T = (_Float16*)(ws + 155648);           // 32*64*2   = 4096 B
    float*    hv   = (float*)(ws + 159744);              // 3*B*64*4  = 6291456 B

    din_prep_kernel<<<(PREP_N4 + 255) / 256, 256, 0, stream>>>(
        mlpW1, mlpW2, attW1, attW2, W1T, W2T, aW1T, aW2T);

    dim3 gatt(B_SZ, 3);
    din_attention_kernel<<<gatt, 256, ATT_SMEM, stream>>>(
        ad, brand, cate, hist_ads, hist_brands, hist_cates,
        E_ad, E_brand, E_cate,
        attW1, attb1, attb2, attW3, attb3,
        aW1T, aW2T, hv);

    din_head_kernel<<<B_SZ / 16, 256, 0, stream>>>(
        user, ad, brand, cate, camp,
        E_user, E_ad, E_brand, E_cate, E_camp,
        hv, W1T, mlpb1, W2T, mlpb2, mlpW3, mlpb3, (float*)d_out);
}
